// BiLSTM_encoder_decoder_7181185319041
// MI455X (gfx1250) — compile-verified, matrix operand order checked by reading, not run on hardware
//
#include <hip/hip_runtime.h>
#include <hip/hip_bf16.h>
#include <math.h>
#include <stdint.h>

// ---------------------------------------------------------------------------
// Types for CDNA5 WMMA (wave32): v_wmma_f32_16x16x32_bf16
// ---------------------------------------------------------------------------
typedef __bf16 bf16_t;
typedef bf16_t v16bf __attribute__((ext_vector_type(16)));
typedef bf16_t v8bf  __attribute__((ext_vector_type(8)));
typedef float  v8f   __attribute__((ext_vector_type(8)));

#define HID   512
#define GATES 2048
#define BATCH 256
#define SEQ   128
#define TLEN  32

#define APAD  8              // LDS row pad: stride 520 bf16 = 260 dwords == 4 mod 64 banks
#define ASTR  (HID + APAD)   // 520
#define XSTR  (1024 + APAD)  // 1032

// 16-byte global -> LDS copy via register relay (global_load_b128 + ds_store_b128).
__device__ __forceinline__ void cp16_g2l(bf16_t* lds, const bf16_t* g) {
  *reinterpret_cast<v8bf*>(lds) = *reinterpret_cast<const v8bf*>(g);
}

// ---------------------------------------------------------------------------
// Load a 16-lane A/B fragment for 16x16x32 bf16 WMMA.
// 16-bit A (16x32) layout: lanes 0-15 -> row M=lane, K = cb..cb+7 and
// cb+16..cb+23 ; lanes 16-31 -> chunk base shifted by 8 (caller folds into cb).
// Works for both LDS (ds_load_b128) and global (global_load_b128) pointers.
// ---------------------------------------------------------------------------
__device__ __forceinline__ v16bf load_frag(const bf16_t* __restrict__ base,
                                           int row, int rowstride, int cb) {
  const bf16_t* p = base + (size_t)row * (size_t)rowstride + cb;
  v8bf a0 = *reinterpret_cast<const v8bf*>(p);
  v8bf a1 = *reinterpret_cast<const v8bf*>(p + 16);
  v16bf r;
#pragma unroll
  for (int i = 0; i < 8; ++i) { r[i] = a0[i]; r[i + 8] = a1[i]; }
  return r;
}

__device__ __forceinline__ float sigf(float x) {
  return 1.0f / (1.0f + __expf(-x));
}

// ---------------------------------------------------------------------------
// Fused LSTM step:  g = hprev @ Whh^T  [+ xin @ Wih^T]  [+ xs @ Wih_s^T] + bias
// then i,f,g,o split -> c' = sig(f)*c + sig(i)*tanh(g); h' = sig(o)*tanh(c').
// Each wave owns a 16(batch) x 16(hidden-unit) tile and computes all 4 gates
// (n, n+512, n+1024, n+1536) so the cell update is wave-local per the C/D
// layout: VGPR r, lane L -> M = r + 8*(L>>4), N = L&15.
// The block's A tiles (hprev rows, optional xin rows) are staged once in LDS
// and shared by all 8 waves; weight (B) tiles stream from L2 with prefetch.
// Grid: 64 blocks x 256 threads (8 waves): block = 1 b-tile x 8 hu-tiles.
// ---------------------------------------------------------------------------
__global__ __launch_bounds__(256)
void lstm_step_kernel(const bf16_t* __restrict__ hprev,   // [256,512] bf16
                      const bf16_t* __restrict__ Whh,     // [2048,512] bf16
                      const bf16_t* __restrict__ xin,     // [256,Kx] bf16 or null
                      const bf16_t* __restrict__ Wih,     // [2048,Kx] bf16 or null
                      int Kx,
                      const float*  __restrict__ xs,      // small input base or null
                      int xs_stride,                      // row stride (floats)
                      const float*  __restrict__ Wih_s,   // [2048,3] fp32 or null
                      const float*  __restrict__ bias,    // [2048] fp32 (bih+bhh)
                      float*        __restrict__ c_state, // [256,512] in/out
                      float*        __restrict__ h_state, // [256,512] out fp32
                      bf16_t*       __restrict__ hbf_out, // [256,512] out bf16
                      bf16_t*       __restrict__ ys,      // null or seq slice base
                      int ys_stride) {
  __shared__ bf16_t sA[16 * ASTR];   // hprev tile: 16 rows x 512 (+pad)
  __shared__ bf16_t sX[16 * XSTR];   // xin tile:   16 rows x <=1024 (+pad)

  const int tid   = threadIdx.x;
  const int wave  = tid >> 5;
  const int lane  = tid & 31;
  const int lhalf = lane >> 4;
  const int l16   = lane & 15;

  const int b0  = (blockIdx.x >> 2) * 16;
  const int hu0 = ((blockIdx.x & 3) * 8 + wave) * 16;

  // ---- cooperative stage of A tiles into LDS (16 threads per row) ----
  {
    const int r = tid >> 4;   // 0..15 row
    const int j = tid & 15;   // 0..15 chunk lane
    const bf16_t* gsrc = hprev + (size_t)(b0 + r) * HID;
    bf16_t* ldst = sA + r * ASTR;
#pragma unroll
    for (int c = 0; c < 4; ++c) {          // 64 chunks/row, 16 threads -> 4 each
      const int e = (j + c * 16) * 8;
      cp16_g2l(ldst + e, gsrc + e);
    }
    if (xin != nullptr) {
      const bf16_t* gx = xin + (size_t)(b0 + r) * Kx;
      bf16_t* lx = sX + r * XSTR;
      const int nchunk = Kx >> 7;          // (Kx/8)/16 chunks per thread
      for (int c = 0; c < nchunk; ++c) {
        const int e = (j + c * 16) * 8;
        cp16_g2l(lx + e, gx + e);
      }
    }
  }
  __syncthreads();

  v8f acc0 = {}, acc1 = {}, acc2 = {}, acc3 = {};

  const int koffA = lhalf * 8;
  const int wrow  = hu0 + l16;

  // ---- recurrent GEMM: h(t-1) @ Whh^T, K = 512 (A from LDS, B from L2) ----
#pragma unroll 4
  for (int k0 = 0; k0 < HID; k0 += 32) {
    const int cb = k0 + koffA;
    // pull next K-slice of the weight stream toward the WGP (global_prefetch_b8)
    if (k0 + 32 < HID)
      __builtin_prefetch(Whh + (size_t)wrow * HID + cb + 32, 0, 1);
    v16bf a  = load_frag(sA, l16, ASTR, cb);
    v16bf w0 = load_frag(Whh, 0 * HID + wrow, HID, cb);
    acc0 = __builtin_amdgcn_wmma_f32_16x16x32_bf16(false, a, false, w0, (short)0, acc0, false, false);
    v16bf w1 = load_frag(Whh, 1 * HID + wrow, HID, cb);
    acc1 = __builtin_amdgcn_wmma_f32_16x16x32_bf16(false, a, false, w1, (short)0, acc1, false, false);
    v16bf w2 = load_frag(Whh, 2 * HID + wrow, HID, cb);
    acc2 = __builtin_amdgcn_wmma_f32_16x16x32_bf16(false, a, false, w2, (short)0, acc2, false, false);
    v16bf w3 = load_frag(Whh, 3 * HID + wrow, HID, cb);
    acc3 = __builtin_amdgcn_wmma_f32_16x16x32_bf16(false, a, false, w3, (short)0, acc3, false, false);
  }

  // ---- optional input GEMM: x(t) @ Wih^T, K = Kx (512 or 1024) ----
  if (xin != nullptr) {
#pragma unroll 2
    for (int k0 = 0; k0 < Kx; k0 += 32) {
      const int cb = k0 + koffA;
      if (k0 + 32 < Kx)
        __builtin_prefetch(Wih + (size_t)wrow * Kx + cb + 32, 0, 1);
      v16bf a  = load_frag(sX, l16, XSTR, cb);
      v16bf w0 = load_frag(Wih, 0 * HID + wrow, Kx, cb);
      acc0 = __builtin_amdgcn_wmma_f32_16x16x32_bf16(false, a, false, w0, (short)0, acc0, false, false);
      v16bf w1 = load_frag(Wih, 1 * HID + wrow, Kx, cb);
      acc1 = __builtin_amdgcn_wmma_f32_16x16x32_bf16(false, a, false, w1, (short)0, acc1, false, false);
      v16bf w2 = load_frag(Wih, 2 * HID + wrow, Kx, cb);
      acc2 = __builtin_amdgcn_wmma_f32_16x16x32_bf16(false, a, false, w2, (short)0, acc2, false, false);
      v16bf w3 = load_frag(Wih, 3 * HID + wrow, Kx, cb);
      acc3 = __builtin_amdgcn_wmma_f32_16x16x32_bf16(false, a, false, w3, (short)0, acc3, false, false);
    }
  }

  const int n  = hu0 + l16;
  const float bi  = bias[0 * HID + n];
  const float bf_ = bias[1 * HID + n];
  const float bg  = bias[2 * HID + n];
  const float bo  = bias[3 * HID + n];

  float wsi0 = 0.f, wsi1 = 0.f, wsi2 = 0.f;
  float wsf0 = 0.f, wsf1 = 0.f, wsf2 = 0.f;
  float wsg0 = 0.f, wsg1 = 0.f, wsg2 = 0.f;
  float wso0 = 0.f, wso1 = 0.f, wso2 = 0.f;
  if (Wih_s != nullptr) {
    wsi0 = Wih_s[(0 * HID + n) * 3 + 0]; wsi1 = Wih_s[(0 * HID + n) * 3 + 1]; wsi2 = Wih_s[(0 * HID + n) * 3 + 2];
    wsf0 = Wih_s[(1 * HID + n) * 3 + 0]; wsf1 = Wih_s[(1 * HID + n) * 3 + 1]; wsf2 = Wih_s[(1 * HID + n) * 3 + 2];
    wsg0 = Wih_s[(2 * HID + n) * 3 + 0]; wsg1 = Wih_s[(2 * HID + n) * 3 + 1]; wsg2 = Wih_s[(2 * HID + n) * 3 + 2];
    wso0 = Wih_s[(3 * HID + n) * 3 + 0]; wso1 = Wih_s[(3 * HID + n) * 3 + 1]; wso2 = Wih_s[(3 * HID + n) * 3 + 2];
  }

#pragma unroll
  for (int r = 0; r < 8; ++r) {
    const int m = b0 + r + lhalf * 8;
    float gi = acc0[r] + bi;
    float gf = acc1[r] + bf_;
    float gg = acc2[r] + bg;
    float go = acc3[r] + bo;
    if (xs != nullptr) {
      const float* xr = xs + (size_t)m * xs_stride;
      const float x0 = xr[0], x1 = xr[1], x2 = xr[2];
      gi += x0 * wsi0 + x1 * wsi1 + x2 * wsi2;
      gf += x0 * wsf0 + x1 * wsf1 + x2 * wsf2;
      gg += x0 * wsg0 + x1 * wsg1 + x2 * wsg2;
      go += x0 * wso0 + x1 * wso1 + x2 * wso2;
    }
    const float ig = sigf(gi);
    const float fg = sigf(gf);
    const float og = sigf(go);
    const float gt = tanhf(gg);
    const size_t idx = (size_t)m * HID + n;
    const float cn = fg * c_state[idx] + ig * gt;
    const float hn = og * tanhf(cn);
    c_state[idx] = cn;
    h_state[idx] = hn;
    hbf_out[idx] = (bf16_t)hn;
    if (ys != nullptr) ys[(size_t)m * ys_stride + n] = (bf16_t)hn;
  }
}

// ---------------------------------------------------------------------------
// Decoder head: out = h3 @ linW^T + lin_b ; update autoregressive input.
// ---------------------------------------------------------------------------
__global__ __launch_bounds__(256)
void dec_output_kernel(const float* __restrict__ h,      // [256,512]
                       const float* __restrict__ linW,   // [512]
                       const float* __restrict__ linb,   // [1]
                       float* __restrict__ dec_inp,      // [256,3] in/out
                       float* __restrict__ out,          // [256,32]
                       int t) {
  const int b = threadIdx.x;
  if (b >= BATCH) return;
  float s = linb[0];
  const float* hr = h + (size_t)b * HID;
#pragma unroll 8
  for (int k = 0; k < HID; ++k) s += hr[k] * linW[k];
  out[b * TLEN + t] = s;
  const float i0 = dec_inp[b * 3 + 0];
  const float i1 = dec_inp[b * 3 + 1];
  const float s1 = i0 - s;
  const float s2 = i1 - s1;
  dec_inp[b * 3 + 0] = s;
  dec_inp[b * 3 + 1] = s1;
  dec_inp[b * 3 + 2] = s2;
}

// ---------------------------------------------------------------------------
// Small utility kernels
// ---------------------------------------------------------------------------
__global__ void f32_to_bf16_kernel(const float* __restrict__ s,
                                   bf16_t* __restrict__ d, int n) {
  int i = blockIdx.x * blockDim.x + threadIdx.x;
  if (i < n) d[i] = (bf16_t)s[i];
}

__global__ void zero_f32_kernel(float* __restrict__ p, int n) {
  int i = blockIdx.x * blockDim.x + threadIdx.x;
  if (i < n) p[i] = 0.0f;
}

__global__ void zero_bf16_kernel(bf16_t* __restrict__ p, int n) {
  int i = blockIdx.x * blockDim.x + threadIdx.x;
  if (i < n) p[i] = (bf16_t)0.0f;
}

__global__ void bias_comb_kernel(const float* __restrict__ a,
                                 const float* __restrict__ b,
                                 float* __restrict__ o, int n) {
  int i = blockIdx.x * blockDim.x + threadIdx.x;
  if (i < n) o[i] = a[i] + b[i];
}

__global__ void dec_inp_init_kernel(const float* __restrict__ x,  // [256,128,3]
                                    float* __restrict__ dec_inp) {
  int i = blockIdx.x * blockDim.x + threadIdx.x;
  if (i < BATCH * 3) {
    int b = i / 3, k = i % 3;
    dec_inp[i] = x[(size_t)b * SEQ * 3 + (SEQ - 1) * 3 + k];
  }
}

// ---------------------------------------------------------------------------
// Host orchestration
// ---------------------------------------------------------------------------
extern "C" void kernel_launch(void* const* d_in, const int* in_sizes, int n_in,
                              void* d_out, int out_size, void* d_ws, size_t ws_size,
                              hipStream_t stream) {
  (void)in_sizes; (void)n_in; (void)out_size; (void)ws_size;

  // Input index map (setup_inputs flattening order):
  // enc cell c (c = layer*2+dir, c<4): [4c+0]=Wih [4c+1]=Whh [4c+2]=bih [4c+3]=bhh
  // dec cell c (c<4):                  [16+4c+0..3]
  // [32]=lin_W [33]=lin_b [34]=x [35]=y [36]=target_len
  const float* x = (const float*)d_in[34];

  // ---- workspace bump allocator ----
  char* ws = (char*)d_ws;
  size_t off = 0;
  auto alloc = [&](size_t bytes) -> char* {
    char* p = ws + off;
    off = (off + bytes + 255) & ~(size_t)255;
    return p;
  };

  bf16_t* whh_bf[8];
  for (int c = 0; c < 8; ++c) whh_bf[c] = (bf16_t*)alloc((size_t)GATES * HID * 2);
  bf16_t* wih1_bf[2];
  for (int d = 0; d < 2; ++d) wih1_bf[d] = (bf16_t*)alloc((size_t)GATES * 1024 * 2);
  bf16_t* wihd_bf[3];
  for (int c = 0; c < 3; ++c) wihd_bf[c] = (bf16_t*)alloc((size_t)GATES * HID * 2);
  float* bias_comb = (float*)alloc((size_t)8 * GATES * 4);
  float* h32[4];
  float* c32[4];
  for (int c = 0; c < 4; ++c) {
    h32[c] = (float*)alloc((size_t)BATCH * HID * 4);
    c32[c] = (float*)alloc((size_t)BATCH * HID * 4);
  }
  bf16_t* hbf[4][2];
  for (int c = 0; c < 4; ++c)
    for (int p = 0; p < 2; ++p) hbf[c][p] = (bf16_t*)alloc((size_t)BATCH * HID * 2);
  bf16_t* ys0 = (bf16_t*)alloc((size_t)SEQ * BATCH * 1024 * 2);  // layer-0 output seq (fwd|bwd)
  float* dec_inp = (float*)alloc((size_t)BATCH * 3 * 4);

  // ---- weight conversion f32 -> bf16 (weights live in L2 afterwards) ----
  auto cvt = [&](const void* src, bf16_t* dst, size_t n) {
    f32_to_bf16_kernel<<<(unsigned)((n + 255) / 256), 256, 0, stream>>>(
        (const float*)src, dst, (int)n);
  };
  for (int c = 0; c < 4; ++c) cvt(d_in[c * 4 + 1], whh_bf[c], (size_t)GATES * HID);
  for (int c = 0; c < 4; ++c) cvt(d_in[16 + c * 4 + 1], whh_bf[4 + c], (size_t)GATES * HID);
  for (int d = 0; d < 2; ++d) cvt(d_in[8 + d * 4 + 0], wih1_bf[d], (size_t)GATES * 1024);
  for (int c = 0; c < 3; ++c) cvt(d_in[16 + (c + 1) * 4 + 0], wihd_bf[c], (size_t)GATES * HID);

  for (int c = 0; c < 4; ++c)
    bias_comb_kernel<<<GATES / 256, 256, 0, stream>>>(
        (const float*)d_in[c * 4 + 2], (const float*)d_in[c * 4 + 3],
        bias_comb + c * GATES, GATES);
  for (int c = 0; c < 4; ++c)
    bias_comb_kernel<<<GATES / 256, 256, 0, stream>>>(
        (const float*)d_in[16 + c * 4 + 2], (const float*)d_in[16 + c * 4 + 3],
        bias_comb + (4 + c) * GATES, GATES);

  // ---- zero initial states ----
  const int stN = BATCH * HID;
  for (int c = 0; c < 4; ++c) {
    zero_f32_kernel<<<(stN + 255) / 256, 256, 0, stream>>>(c32[c], stN);
    zero_f32_kernel<<<(stN + 255) / 256, 256, 0, stream>>>(h32[c], stN);
    zero_bf16_kernel<<<(stN + 255) / 256, 256, 0, stream>>>(hbf[c][0], stN);
  }
  dec_inp_init_kernel<<<3, 256, 0, stream>>>(x, dec_inp);

  const dim3 grid(64), blk(256);

  // ---- encoder: 2 layers x {fwd,bwd} x 128 steps ----
  for (int layer = 0; layer < 2; ++layer) {
    for (int dir = 0; dir < 2; ++dir) {
      const int cell = layer * 2 + dir;
      for (int s = 0; s < SEQ; ++s) {
        const int t = dir ? (SEQ - 1 - s) : s;
        const bf16_t* hprev = hbf[cell][s & 1];
        bf16_t* hout = hbf[cell][(s + 1) & 1];
        if (layer == 0) {
          lstm_step_kernel<<<grid, blk, 0, stream>>>(
              hprev, whh_bf[cell],
              nullptr, nullptr, 0,
              x + t * 3, SEQ * 3, (const float*)d_in[cell * 4 + 0],
              bias_comb + cell * GATES,
              c32[cell], h32[cell], hout,
              ys0 + (size_t)t * BATCH * 1024 + dir * HID, 1024);
        } else {
          lstm_step_kernel<<<grid, blk, 0, stream>>>(
              hprev, whh_bf[cell],
              ys0 + (size_t)t * BATCH * 1024, wih1_bf[dir], 1024,
              nullptr, 0, nullptr,
              bias_comb + cell * GATES,
              c32[cell], h32[cell], hout,
              nullptr, 0);
        }
      }
    }
  }
  // After 128 steps every cell's final bf16 h sits in parity-0 buffer, and
  // h32/c32 hold final states in the hns order (l0f,l0b,l1f,l1b) == dec cells.

  // ---- decoder: 32 steps x 4 stacked cells + linear head ----
  float* out = (float*)d_out;
  for (int s = 0; s < TLEN; ++s) {
    for (int c = 0; c < 4; ++c) {
      const bf16_t* hprev = hbf[c][s & 1];
      bf16_t* hout = hbf[c][(s + 1) & 1];
      if (c == 0) {
        lstm_step_kernel<<<grid, blk, 0, stream>>>(
            hprev, whh_bf[4],
            nullptr, nullptr, 0,
            dec_inp, 3, (const float*)d_in[16 + 0],
            bias_comb + 4 * GATES,
            c32[0], h32[0], hout, nullptr, 0);
      } else {
        lstm_step_kernel<<<grid, blk, 0, stream>>>(
            hprev, whh_bf[4 + c],
            hbf[c - 1][(s + 1) & 1], wihd_bf[c - 1], HID,
            nullptr, 0, nullptr,
            bias_comb + (4 + c) * GATES,
            c32[c], h32[c], hout, nullptr, 0);
      }
    }
    dec_output_kernel<<<1, 256, 0, stream>>>(
        h32[3], (const float*)d_in[32], (const float*)d_in[33], dec_inp, out, s);
  }
}
